// PCCASimple_45226005627432
// MI455X (gfx1250) — matrix operand order repaired and checked by reading, not compiled, not run on hardware
//
#include <hip/hip_runtime.h>
#include <cstddef>

// ---------------------------------------------------------------------------
// Types for CDNA5 WMMA (wave32): V_WMMA_F32_16X16X32_F16
// ---------------------------------------------------------------------------
typedef __attribute__((ext_vector_type(16))) _Float16 v16h;
typedef __attribute__((ext_vector_type(8)))  _Float16 v8h;
typedef __attribute__((ext_vector_type(4)))  _Float16 v4h;
typedef __attribute__((ext_vector_type(8)))  float    v8f;

static __device__ __forceinline__ v8f zero8() {
  v8f z = {0.f, 0.f, 0.f, 0.f, 0.f, 0.f, 0.f, 0.f};
  return z;
}

static __device__ __forceinline__ v8f wmma16(v16h a, v16h b, v8f c) {
  return __builtin_amdgcn_wmma_f32_16x16x32_f16(
      /*neg_a=*/false, a, /*neg_b=*/false, b,
      /*c_mod=*/(short)0, c, /*reuse_a=*/false, /*reuse_b=*/false);
}

// A-operand fragment (16x32, MxK) from row-major LDS matrix (stride ld halves).
// Lane layout (ISA 7.12.2): m = lane&15, half = lane>>4;
//   a[e] = A[m][k0 + 8*half + e]          for e in 0..7
//   a[e] = A[m][k0 + 16 + 8*half + (e-8)] for e in 8..15
static __device__ __forceinline__ v16h load_a(const _Float16* base, int ld,
                                              int m0, int k0, int l15, int h) {
  const _Float16* row = base + (m0 + l15) * ld + k0 + h * 8;
  v8h p0 = *(const v8h*)row;
  v8h p1 = *(const v8h*)(row + 16);
  return __builtin_shufflevector(p0, p1, 0, 1, 2, 3, 4, 5, 6, 7, 8, 9, 10, 11,
                                 12, 13, 14, 15);
}

// B-operand fragment (32x16, KxN) where logical B[k][n] = BT[n][k] and BT is
// row-major in LDS with stride ld. Lane layout: n = lane&15, half = lane>>4;
//   b[e] = BT[n0+n][k0 + 8*half + e]            for e in 0..7
//   b[e] = BT[n0+n][k0 + 16 + 8*half + (e-8)]   for e in 8..15
static __device__ __forceinline__ v16h load_bt(const _Float16* baseT, int ld,
                                               int k0, int n0, int l15, int h) {
  const _Float16* row = baseT + (n0 + l15) * ld + k0 + h * 8;
  v8h p0 = *(const v8h*)row;
  v8h p1 = *(const v8h*)(row + 16);
  return __builtin_shufflevector(p0, p1, 0, 1, 2, 3, 4, 5, 6, 7, 8, 9, 10, 11,
                                 12, 13, 14, 15);
}

// Guarded load of 4 consecutive columns as f16 (overloaded on source type).
static __device__ __forceinline__ void load4cols(const float* __restrict__ row,
                                                 int c, int n, _Float16* o) {
  if (c + 3 < n) {
    float4 v = *(const float4*)(row + c);
    o[0] = (_Float16)v.x; o[1] = (_Float16)v.y;
    o[2] = (_Float16)v.z; o[3] = (_Float16)v.w;
  } else {
    o[0] = (c + 0 < n) ? (_Float16)row[c + 0] : (_Float16)0.f;
    o[1] = (c + 1 < n) ? (_Float16)row[c + 1] : (_Float16)0.f;
    o[2] = (c + 2 < n) ? (_Float16)row[c + 2] : (_Float16)0.f;
    o[3] = (_Float16)0.f;
  }
}
static __device__ __forceinline__ void load4cols(const _Float16* __restrict__ row,
                                                 int c, int n, _Float16* o) {
  if (c + 3 < n) {
    v4h v = *(const v4h*)(row + c);
    o[0] = v[0]; o[1] = v[1]; o[2] = v[2]; o[3] = v[3];
  } else {
    o[0] = (c + 0 < n) ? row[c + 0] : (_Float16)0.f;
    o[1] = (c + 1 < n) ? row[c + 1] : (_Float16)0.f;
    o[2] = (c + 2 < n) ? row[c + 2] : (_Float16)0.f;
    o[3] = (_Float16)0.f;
  }
}

#define STR 136  // LDS row stride in halves (272B: 16B aligned, bank-spread)

// ---------------------------------------------------------------------------
// Streaming pass: for a 128-column tile of y (512 x n):
//   Ez   = beta(64x512) @ Ytile(512x128)          (WMMA, f32 accum)
//   Ezz += Ez @ Ez^T                               (WMMA)
//   Syz += Ytile @ Ez^T                            (WMMA)
// Partials accumulated per-block (deterministic), reduced by fa_reduce.
// T = float (reads original y) or _Float16 (reads pre-converted copy).
// ---------------------------------------------------------------------------
template <typename T>
__global__ __launch_bounds__(256) void fa_stream_t(
    const T* __restrict__ ysrc, const _Float16* __restrict__ betaH,
    float* __restrict__ part, int n, int nTiles) {
  __shared__ alignas(16) _Float16 sY[128 * STR];  // Y chunk (col-major, then row-major)
  __shared__ alignas(16) _Float16 sE[64 * STR];   // beta chunk, then Ez (row-major)

  const int tid = threadIdx.x;
  const int lane = tid & 31;
  const int wave = tid >> 5;       // 0..7
  const int l15 = lane & 15;
  const int h = (lane >> 4) & 1;

  float* myS = part + (size_t)blockIdx.x * 36864;  // 512*64 Syz partial
  float* myE = myS + 32768;                        // 64*64  Ezz partial
  for (int i = tid; i < 36864; i += 256) myS[i] = 0.0f;

  for (int tile = blockIdx.x; tile < nTiles; tile += gridDim.x) {
    const int c0 = tile * 128;

    // ---------------- Phase 1: Ez = beta @ Ytile --------------------------
    v8f ez0 = zero8(), ez1 = zero8(), ez2 = zero8(), ez3 = zero8();
    for (int ch = 0; ch < 4; ++ch) {  // p-chunks of 128 rows (K = 512 total)
      __syncthreads();
      // beta chunk row-major: sE[a][k] = beta[a][ch*128+k]
      for (int idx = tid; idx < 64 * 16; idx += 256) {
        int a = idx >> 4;
        int k8 = (idx & 15) << 3;
        *(v8h*)(sE + a * STR + k8) =
            *(const v8h*)(betaH + a * 512 + ch * 128 + k8);
      }
      // Y chunk column-major: sY[c][r] = y[ch*128+r][c0+c]
      for (int idx = tid; idx < 128 * 32; idx += 256) {
        int r = idx >> 5;
        int c4 = (idx & 31) << 2;
        int j = ch * 128 + r;
        int c = c0 + c4;
        const T* row = ysrc + (size_t)j * n;
        _Float16 xv[4];
        load4cols(row, c, n, xv);
        if (j + 128 < 512) __builtin_prefetch(ysrc + (size_t)(j + 128) * n + c, 0, 1);
        sY[(c4 + 0) * STR + r] = xv[0];
        sY[(c4 + 1) * STR + r] = xv[1];
        sY[(c4 + 2) * STR + r] = xv[2];
        sY[(c4 + 3) * STR + r] = xv[3];
      }
      __syncthreads();
      // wave w owns output columns [16w,16w+16); all 4 M-tiles of Ez
      for (int kt = 0; kt < 4; ++kt) {
        v16h b = load_bt(sY, STR, kt * 32, wave * 16, l15, h);
        ez0 = wmma16(load_a(sE, STR, 0, kt * 32, l15, h), b, ez0);
        ez1 = wmma16(load_a(sE, STR, 16, kt * 32, l15, h), b, ez1);
        ez2 = wmma16(load_a(sE, STR, 32, kt * 32, l15, h), b, ez2);
        ez3 = wmma16(load_a(sE, STR, 48, kt * 32, l15, h), b, ez3);
      }
    }
    __syncthreads();
    // Store Ez (64 x 128) to sE row-major as f16
    {
      int col = wave * 16 + l15;
      for (int r = 0; r < 8; ++r) {
        sE[(0 + r + 8 * h) * STR + col] = (_Float16)ez0[r];
        sE[(16 + r + 8 * h) * STR + col] = (_Float16)ez1[r];
        sE[(32 + r + 8 * h) * STR + col] = (_Float16)ez2[r];
        sE[(48 + r + 8 * h) * STR + col] = (_Float16)ez3[r];
      }
    }
    __syncthreads();

    // ---------------- GEMM2: Ezz += Ez @ Ez^T -----------------------------
    for (int t2 = 0; t2 < 2; ++t2) {
      int tileid = wave * 2 + t2;        // 16 tiles: 4x4
      int mt = tileid >> 2, nt = tileid & 3;
      v8f acc = zero8();
      for (int kt = 0; kt < 4; ++kt) {
        v16h a = load_a(sE, STR, mt * 16, kt * 32, l15, h);
        v16h b = load_bt(sE, STR, kt * 32, nt * 16, l15, h);
        acc = wmma16(a, b, acc);
      }
      int col = nt * 16 + l15;
      for (int r = 0; r < 8; ++r) {
        int row = mt * 16 + r + 8 * h;
        myE[row * 64 + col] += acc[r];
      }
    }

    // ---------------- Phase 3: Syz += Ytile @ Ez^T ------------------------
    for (int ch = 0; ch < 4; ++ch) {
      __syncthreads();
      // Y chunk row-major: sY[r][c] = y[ch*128+r][c0+c]
      for (int idx = tid; idx < 128 * 32; idx += 256) {
        int r = idx >> 5;
        int c4 = (idx & 31) << 2;
        int j = ch * 128 + r;
        int c = c0 + c4;
        const T* row = ysrc + (size_t)j * n;
        _Float16 xv[4];
        load4cols(row, c, n, xv);
        if (j + 128 < 512) __builtin_prefetch(ysrc + (size_t)(j + 128) * n + c, 0, 1);
        v4h pk = {xv[0], xv[1], xv[2], xv[3]};
        *(v4h*)(sY + r * STR + c4) = pk;
      }
      __syncthreads();
      // wave w owns rows [16w,16w+16) of the chunk; N-tiles 0..3
      v8f a0 = zero8(), a1 = zero8(), a2 = zero8(), a3 = zero8();
      for (int kt = 0; kt < 4; ++kt) {
        v16h a = load_a(sY, STR, wave * 16, kt * 32, l15, h);
        a0 = wmma16(a, load_bt(sE, STR, kt * 32, 0, l15, h), a0);
        a1 = wmma16(a, load_bt(sE, STR, kt * 32, 16, l15, h), a1);
        a2 = wmma16(a, load_bt(sE, STR, kt * 32, 32, l15, h), a2);
        a3 = wmma16(a, load_bt(sE, STR, kt * 32, 48, l15, h), a3);
      }
      for (int r = 0; r < 8; ++r) {
        int prow = ch * 128 + wave * 16 + r + 8 * h;
        float* rowp = myS + prow * 64;
        rowp[0 + l15]  += a0[r];
        rowp[16 + l15] += a1[r];
        rowp[32 + l15] += a2[r];
        rowp[48 + l15] += a3[r];
      }
    }
  }
}

// ---------------------------------------------------------------------------
// One-time f32 -> f16 conversion of y into workspace
// ---------------------------------------------------------------------------
__global__ void fa_half(const float* __restrict__ y, _Float16* __restrict__ yH,
                        long total) {
  long i = ((long)blockIdx.x * 256 + threadIdx.x) * 4;
  if (i + 3 < total) {
    float4 v = *(const float4*)(y + i);
    v4h o = {(_Float16)v.x, (_Float16)v.y, (_Float16)v.z, (_Float16)v.w};
    *(v4h*)(yH + i) = o;
  } else if (i < total) {
    for (long t = i; t < total; ++t) yH[t] = (_Float16)y[t];
  }
}

// ---------------------------------------------------------------------------
// Deterministic reduction of per-block partials
// ---------------------------------------------------------------------------
__global__ void fa_reduce(const float* __restrict__ part, float* __restrict__ Syz,
                          float* __restrict__ Ezz, int nblk) {
  int i = blockIdx.x * 256 + threadIdx.x;
  if (i < 32768) {
    float s = 0.f;
    for (int b = 0; b < nblk; ++b) s += part[(size_t)b * 36864 + i];
    Syz[i] = s;
  } else if (i < 36864) {
    int j = i - 32768;
    float s = 0.f;
    for (int b = 0; b < nblk; ++b) s += part[(size_t)b * 36864 + 32768 + j];
    Ezz[j] = s;
  }
}

// ---------------------------------------------------------------------------
// Small dense kernels (f32)
// ---------------------------------------------------------------------------
__global__ void fa_init(const float* __restrict__ L1, const float* __restrict__ L2,
                        const float* __restrict__ P1, const float* __restrict__ P2,
                        float* __restrict__ L, float* __restrict__ Psi) {
  int i = blockIdx.x * 256 + threadIdx.x;  // 16384 threads
  L[i] = L1[i];
  L[16384 + i] = L2[i];
  if (i < 256) { Psi[i] = P1[i]; Psi[256 + i] = P2[i]; }
}

__global__ __launch_bounds__(256) void fa_yy(const float* __restrict__ y,
                                             float* __restrict__ yy, int n) {
  __shared__ float red[256];
  int r = blockIdx.x;
  const float* row = y + (size_t)r * n;
  float s = 0.f;
  for (int c = threadIdx.x; c < n; c += 256) { float v = row[c]; s += v * v; }
  red[threadIdx.x] = s;
  __syncthreads();
  for (int off = 128; off > 0; off >>= 1) {
    if (threadIdx.x < off) red[threadIdx.x] += red[threadIdx.x + off];
    __syncthreads();
  }
  if (threadIdx.x == 0) yy[r] = red[0];
}

// G = L^T diag(1/Psi) L   (64x64)
__global__ void fa_g(const float* __restrict__ L, const float* __restrict__ Psi,
                     float* __restrict__ G) {
  int a = blockIdx.x, b = threadIdx.x;
  if (b < 64) {
    float s = 0.f;
    for (int j = 0; j < 512; ++j) s += L[j * 64 + a] * L[j * 64 + b] / Psi[j];
    G[a * 64 + b] = s;
  }
}

// mode 0: invert (I + A);  mode 1: invert (fn*(I - A) + B).  Gauss-Jordan.
__global__ __launch_bounds__(256) void fa_inv64(const float* __restrict__ A,
                                                const float* __restrict__ B,
                                                float fn, int mode,
                                                float* __restrict__ Out) {
  __shared__ float Wm[64 * 130];
  __shared__ float mult[64];
  int tid = threadIdx.x;
  for (int idx = tid; idx < 64 * 128; idx += 256) {
    int r = idx >> 7, c = idx & 127;
    float v;
    if (c < 64) {
      float diag = (r == c) ? 1.f : 0.f;
      v = (mode == 0) ? (diag + A[r * 64 + c])
                      : (fn * (diag - A[r * 64 + c]) + B[r * 64 + c]);
    } else {
      v = ((c - 64) == r) ? 1.f : 0.f;
    }
    Wm[r * 130 + c] = v;
  }
  for (int p = 0; p < 64; ++p) {
    __syncthreads();
    float ip = 1.f / Wm[p * 130 + p];
    __syncthreads();
    if (tid < 128) Wm[p * 130 + tid] *= ip;
    if (tid < 64) mult[tid] = (tid == p) ? 0.f : Wm[tid * 130 + p];
    __syncthreads();
    for (int idx = tid; idx < 64 * 128; idx += 256) {
      int r = idx >> 7, c = idx & 127;
      Wm[r * 130 + c] -= mult[r] * Wm[p * 130 + c];
    }
  }
  __syncthreads();
  for (int idx = tid; idx < 64 * 64; idx += 256) {
    int r = idx >> 6, c = idx & 63;
    Out[r * 64 + c] = Wm[r * 130 + 64 + c];
  }
}

// beta[a][j] = (1/Psi[j]) * sum_b Minv[a][b] * L[j][b];  f32 + f16 copies
__global__ void fa_beta(const float* __restrict__ Minv, const float* __restrict__ L,
                        const float* __restrict__ Psi, float* __restrict__ betaF,
                        _Float16* __restrict__ betaH) {
  int a = threadIdx.x & 63;
  int j = blockIdx.x * 4 + (threadIdx.x >> 6);
  float s = 0.f;
  for (int b = 0; b < 64; ++b) s += Minv[a * 64 + b] * L[j * 64 + b];
  s /= Psi[j];
  betaF[a * 512 + j] = s;
  betaH[a * 512 + j] = (_Float16)s;
}

// betaL = beta @ L  (64x64)
__global__ void fa_betaL(const float* __restrict__ betaF, const float* __restrict__ L,
                         float* __restrict__ betaL) {
  int a = blockIdx.x, b = threadIdx.x;
  if (b < 64) {
    float s = 0.f;
    for (int j = 0; j < 512; ++j) s += betaF[a * 512 + j] * L[j * 64 + b];
    betaL[a * 64 + b] = s;
  }
}

// Lnew = Syz @ SzzInv;  Psi[j] = (yy[j] - sum_a Lnew[j][a]*Syz[j][a]) / n
__global__ __launch_bounds__(256) void fa_em(const float* __restrict__ Syz,
                                             const float* __restrict__ SzzInv,
                                             const float* __restrict__ yy, float fn,
                                             float* __restrict__ L,
                                             float* __restrict__ Psi) {
  __shared__ float red[256];
  int a = threadIdx.x & 63;
  int jl = threadIdx.x >> 6;
  int j = blockIdx.x * 4 + jl;
  float s = 0.f;
  for (int b = 0; b < 64; ++b) s += Syz[j * 64 + b] * SzzInv[b * 64 + a];
  L[j * 64 + a] = s;
  red[threadIdx.x] = s * Syz[j * 64 + a];
  __syncthreads();
  if (a == 0) {
    float d = 0.f;
    for (int t = 0; t < 64; ++t) d += red[jl * 64 + t];
    Psi[j] = (yy[j] - d) / fn;
  }
}

// nll = 0.5*n*sum(log Psi) + 0.5*sum(yy/Psi) - sum_j (1/Psi_j) <L_j, Syz_j>
//       + 0.5*tr(G @ Szz),  Szz = n*(I - betaL) + Ezz
__global__ __launch_bounds__(256) void fa_nll(
    const float* __restrict__ L, const float* __restrict__ Psi,
    const float* __restrict__ G, const float* __restrict__ betaL,
    const float* __restrict__ Syz, const float* __restrict__ Ezz,
    const float* __restrict__ yy, float fn, float* __restrict__ out) {
  __shared__ float red[256];
  int tid = threadIdx.x;
  float part = 0.f;
  for (int j = tid; j < 512; j += 256) {
    float pi = 1.f / Psi[j];
    float dot = 0.f;
    for (int k2 = 0; k2 < 64; ++k2) dot += L[j * 64 + k2] * Syz[j * 64 + k2];
    part += 0.5f * pi * yy[j] - pi * dot + 0.5f * fn * logf(Psi[j]);
  }
  for (int idx = tid; idx < 4096; idx += 256) {
    int r = idx >> 6, c = idx & 63;
    float szz = fn * (((r == c) ? 1.f : 0.f) - betaL[r * 64 + c]) + Ezz[r * 64 + c];
    part += 0.5f * G[c * 64 + r] * szz;
  }
  red[tid] = part;
  __syncthreads();
  for (int off = 128; off > 0; off >>= 1) {
    if (tid < off) red[tid] += red[tid + off];
    __syncthreads();
  }
  if (tid == 0) out[0] = red[0];
}

__global__ void fa_finish(const float* __restrict__ L, const float* __restrict__ Psi,
                          float* __restrict__ out) {
  int i = blockIdx.x * 256 + threadIdx.x;
  if (i < 32768) out[i] = L[i];
  else if (i < 33280) out[i] = Psi[i - 32768];
}

// ---------------------------------------------------------------------------
// Host orchestration
// ---------------------------------------------------------------------------
extern "C" void kernel_launch(void* const* d_in, const int* in_sizes, int n_in,
                              void* d_out, int out_size, void* d_ws, size_t ws_size,
                              hipStream_t stream) {
  (void)n_in; (void)out_size;
  const float* y  = (const float*)d_in[0];
  const float* L1 = (const float*)d_in[1];
  const float* L2 = (const float*)d_in[2];
  const float* P1 = (const float*)d_in[3];
  const float* P2 = (const float*)d_in[4];
  const int n = in_sizes[0] / 512;

  float* W = (float*)d_ws;
  float* L      = W + 0;        // 512*64
  float* Psi    = W + 32768;    // 512
  float* yy     = W + 33280;    // 512
  float* G      = W + 33792;    // 64*64
  float* Minv   = W + 37888;    // 64*64
  float* betaF  = W + 41984;    // 64*512
  float* betaL  = W + 74752;    // 64*64
  float* Syz    = W + 78848;    // 512*64
  float* Ezz    = W + 111616;   // 64*64
  float* SzzInv = W + 115712;   // 64*64
  _Float16* betaH = (_Float16*)(W + 119808);  // 64*512 halves
  float* out    = (float*)d_out;

  long availF = (long)(ws_size / 4) - 136192L;
  long total  = (long)512 * n;
  long yhF    = ((total + 1) / 2 + 3) & ~3L;  // floats for f16 copy, 16B aligned

  bool useHalf = false;
  _Float16* yH = nullptr;
  float* part;
  int nblk;
  if (availF >= yhF + 36864L) {
    useHalf = true;
    yH = (_Float16*)(W + 136192);
    part = W + 136192 + yhF;
    nblk = (int)((availF - yhF) / 36864L);
  } else {
    part = W + 136192;
    nblk = (int)(availF / 36864L);
  }
  if (nblk < 1) nblk = 1;
  if (nblk > 256) nblk = 256;
  int nTiles = (n + 127) / 128;
  if (nblk > nTiles) nblk = nTiles;

  fa_init<<<64, 256, 0, stream>>>(L1, L2, P1, P2, L, Psi);
  fa_yy<<<512, 256, 0, stream>>>(y, yy, n);
  if (useHalf) {
    int cvtBlocks = (int)((total / 4 + 255) / 256) + 1;
    fa_half<<<cvtBlocks, 256, 0, stream>>>(y, yH, total);
  }

  auto pass = [&]() {
    fa_g<<<64, 64, 0, stream>>>(L, Psi, G);
    fa_inv64<<<1, 256, 0, stream>>>(G, G, 0.f, 0, Minv);
    fa_beta<<<128, 256, 0, stream>>>(Minv, L, Psi, betaF, betaH);
    fa_betaL<<<64, 64, 0, stream>>>(betaF, L, betaL);
    if (useHalf)
      fa_stream_t<_Float16><<<nblk, 256, 0, stream>>>(yH, betaH, part, n, nTiles);
    else
      fa_stream_t<float><<<nblk, 256, 0, stream>>>(y, betaH, part, n, nTiles);
    fa_reduce<<<144, 256, 0, stream>>>(part, Syz, Ezz, nblk);
  };

  pass();  // moments for the first E-step
  for (int t = 0; t < 8; ++t) {
    fa_inv64<<<1, 256, 0, stream>>>(betaL, Ezz, (float)n, 1, SzzInv);
    fa_em<<<128, 256, 0, stream>>>(Syz, SzzInv, yy, (float)n, L, Psi);
    pass();  // moments under updated params: used for nll(t) AND next E-step
    fa_nll<<<1, 256, 0, stream>>>(L, Psi, G, betaL, Syz, Ezz, yy, (float)n,
                                  out + 33280 + t);
  }
  fa_finish<<<131, 256, 0, stream>>>(L, Psi, out);
}
